// VGG16_3DNetwork_52518860096072
// MI455X (gfx1250) — matrix operand order repaired and checked by reading, not compiled, not run on hardware
//
#include <hip/hip_runtime.h>

// ---------------------------------------------------------------------------
// MI455X (gfx1250) VGG-3D forward.
// Conv layers: implicit GEMM with V_WMMA_F32_16X16X32_F16 (wave32).
//   M = 16 consecutive output voxels along innermost spatial dim (NDHWC rows)
//   N = 16 output channels, K = 27 taps x Cin, chunked by 32 (f16).
// Interior A-panels are staged with GLOBAL_LOAD_ASYNC_TO_LDS_B128 (ASYNCcnt);
// border tiles fall back to predicated scalar staging (zero padding).
// B-panels are fetched with 128-bit loads and stored transposed so each wave
// reads its WMMA B fragment as two ds_load_b128.
// BatchNorm (training-mode): conv -> f32 ws, per-channel sum/sumsq reduce,
// fused BN(+ReLU)(+maxpool2) writing f16 activations for the next layer.
// ---------------------------------------------------------------------------

typedef __attribute__((ext_vector_type(16))) _Float16 v16h;
typedef __attribute__((ext_vector_type(8)))  _Float16 v8h;
typedef __attribute__((ext_vector_type(8)))  float    v8f;

#define BT_STRIDE 40   // f16 per row of transposed-B LDS panel (80B, 16B-aligned)

// ---------------------------------------------------------------------------
__global__ void zero_f32_k(float* __restrict__ p, int n) {
  int i = blockIdx.x * 256 + threadIdx.x;
  if (i < n) p[i] = 0.0f;
}

__global__ void f32_to_f16_k(const float* __restrict__ in, _Float16* __restrict__ out,
                             int n) {
  int i = blockIdx.x * 256 + threadIdx.x;
  if (i < n) out[i] = (_Float16)in[i];
}

// ---------------------------------------------------------------------------
// Implicit-GEMM conv3d 3x3x3 SAME, stride 1.
// act:  f16 [2, D, D, D, CIN]      wgt: f16 [27, CIN, COUT]  (tap-major)
// outc: f32 [2, D, D, D, COUT]
// grid: (ceil(D/16), D*D, 2), block 256 (8 waves).
template <int CIN, int COUT>
__global__ __launch_bounds__(256)
void conv3d_wmma_k(const _Float16* __restrict__ act, const _Float16* __restrict__ wgt,
                   float* __restrict__ outc, int D) {
  constexpr int TILESN = COUT / 16;             // 4, 8 or 16
  constexpr int CHUNKS = (CIN + 31) / 32;       // 1, 2, 4 or 8
  constexpr int KV     = (CIN < 32) ? CIN : 32; // valid K rows per chunk
  constexpr int BITER  = (32 * COUT) / (256 * 8); // 128-bit B staging iters

  __shared__ _Float16 ldsA[16 * 32];
  __shared__ _Float16 ldsBT[COUT * BT_STRIDE];

  const int tid  = threadIdx.x;
  const int lane = tid & 31;
  const int wave = tid >> 5;
  const int b  = blockIdx.z;
  const int x  = blockIdx.y / D;
  const int y  = blockIdx.y % D;
  const int z0 = blockIdx.x * 16;

  const bool has0 = (TILESN >= 8) ? true : (wave < TILESN);  // wave-uniform

  v8f acc0, acc1;
#pragma unroll
  for (int r = 0; r < 8; ++r) { acc0[r] = 0.0f; acc1[r] = 0.0f; }

  const int mrow = lane & 15;
  const int hi   = lane >> 4;                   // lane half selects K sub-blocks

  for (int tap = 0; tap < 27; ++tap) {
    const int ax = tap / 9, ay = (tap / 3) % 3, az = tap % 3;
    const int xi = x + ax - 1, yi = y + ay - 1;
    const bool xyok = ((unsigned)xi < (unsigned)D) && ((unsigned)yi < (unsigned)D);
    // all 16 z-rows in range -> async path legal (block-uniform condition)
    const bool zfull = (z0 + az - 1 >= 0) && (z0 + az + 14 < D);
    // L2-resident weights; hint next tap's panel (global_prefetch_b8)
    if (tap < 26) __builtin_prefetch(wgt + (tap + 1) * CIN * COUT, 0, 0);

    for (int ch = 0; ch < CHUNKS; ++ch) {
      bool async_a = false;
      // ---- stage A panel: 16 voxels x 32 f16 ------------------------------
      if constexpr (KV == 32) {
        if (xyok && zfull) {
          // Interior tile: async DMA 16 rows x 64B into LDS, no VGPR transit.
          async_a = true;
          if (tid < 64) {                       // waves 0-1, EXEC all ones
            const int row  = tid >> 2;          // voxel row 0..15
            const int part = tid & 3;           // 16B quarter of the row
            const int elt  = (((b * D + xi) * D + yi) * D + (z0 + row + az - 1)) * CIN
                             + (ch << 5);
            const unsigned goff = (unsigned)elt * 2u + (unsigned)part * 16u;
            const unsigned loff = (unsigned)(size_t)ldsA
                                  + (unsigned)(row * 64 + part * 16);
            asm volatile("global_load_async_to_lds_b128 %0, %1, %2 offset:0"
                         :: "v"(loff), "v"(goff),
                            "s"((unsigned long long)(size_t)act)
                         : "memory");
          }
        }
      }
      if (!async_a) {
        // Border (or CIN<32) tile: predicated scalar staging with zero pad.
        const int r  = tid >> 4;                // voxel row 0..15
        const int cc = (tid & 15) * 2;          // channel pair
        const int zi = z0 + r + az - 1;
        _Float16 v0 = (_Float16)0.0f, v1 = (_Float16)0.0f;
        if (xyok && (unsigned)zi < (unsigned)D) {
          const int base = (((b * D + xi) * D + yi) * D + zi) * CIN + (ch << 5);
          if constexpr (KV == 32) {
            v0 = act[base + cc];
            v1 = act[base + cc + 1];
          } else {
            if (cc < KV)     v0 = act[base + cc];
            if (cc + 1 < KV) v1 = act[base + cc + 1];
          }
        }
        ldsA[r * 32 + cc]     = v0;
        ldsA[r * 32 + cc + 1] = v1;
      }
      // ---- stage B panel (transposed: [n][k]), 128-bit global loads --------
      {
        const int wbase = (tap * CIN + (ch << 5)) * COUT;
#pragma unroll
        for (int it = 0; it < BITER; ++it) {
          const int p  = tid + it * 256;        // v8h vector index
          const int k  = (p * 8) / COUT;        // B row (K)
          const int n0 = (p * 8) % COUT;        // first of 8 output channels
          v8h wv;
#pragma unroll
          for (int j = 0; j < 8; ++j) wv[j] = (_Float16)0.0f;
          if (k < KV) wv = *(const v8h*)(wgt + wbase + k * COUT + n0);
#pragma unroll
          for (int j = 0; j < 8; ++j) ldsBT[(n0 + j) * BT_STRIDE + k] = wv[j];
        }
      }
      if (async_a && tid < 64)
        asm volatile("s_wait_asynccnt 0x0" ::: "memory");
      __syncthreads();

      // ---- A fragment, contiguous K order (same as B fragment) ------------------------
      // lane<16 : elems 0-15 -> K 0..15
      // lane>=16: elems 0-15 -> K 16..31
      v16h afrag;
      {
        const _Float16* pa = ldsA + mrow * 32 + hi * 16;
#pragma unroll
        for (int e = 0; e < 16; ++e) afrag[e] = pa[e];
      }
      // ---- WMMA accumulate: fixed registers, wave-uniform predication ------
      if (has0) {
        const _Float16* pb = ldsBT + (wave * 16 + mrow) * BT_STRIDE + hi * 16;
        v16h bfrag;
#pragma unroll
        for (int e = 0; e < 16; ++e) bfrag[e] = pb[e];
        acc0 = __builtin_amdgcn_wmma_f32_16x16x32_f16(
            false, afrag, false, bfrag, (short)0, acc0, false, false);
      }
      if constexpr (TILESN > 8) {
        const _Float16* pb = ldsBT + ((wave + 8) * 16 + mrow) * BT_STRIDE + hi * 16;
        v16h bfrag;
#pragma unroll
        for (int e = 0; e < 16; ++e) bfrag[e] = pb[e];
        acc1 = __builtin_amdgcn_wmma_f32_16x16x32_f16(
            false, afrag, false, bfrag, (short)0, acc1, false, false);
      }
      __syncthreads();
    }
  }

  // ---- store D fragments: VGPR r -> voxel M = r + 8*hi, chan N = tile*16+mrow
  const int obase = (((b * D + x) * D + y) * D) * COUT;
  if (has0) {
    const int ncol = wave * 16 + mrow;
#pragma unroll
    for (int r = 0; r < 8; ++r) {
      const int z = z0 + hi * 8 + r;
      if (z < D) outc[obase + z * COUT + ncol] = acc0[r];
    }
  }
  if constexpr (TILESN > 8) {
    const int ncol = (wave + 8) * 16 + mrow;
#pragma unroll
    for (int r = 0; r < 8; ++r) {
      const int z = z0 + hi * 8 + r;
      if (z < D) outc[obase + z * COUT + ncol] = acc1[r];
    }
  }
}

// ---------------------------------------------------------------------------
// Per-channel sum / sumsq over all voxels (training-mode BN statistics).
__global__ void channel_stats_k(const float* __restrict__ x, float* __restrict__ sum,
                                float* __restrict__ sumsq, int nvox, int C) {
  __shared__ float s1[256];
  __shared__ float s2[256];
  const int tid = threadIdx.x;
  const int c = tid % C;
  const int g = tid / C;
  const int groups = 256 / C;
  int per = (nvox + gridDim.x - 1) / gridDim.x;
  int v0 = blockIdx.x * per;
  int v1 = v0 + per; if (v1 > nvox) v1 = nvox;
  float a = 0.0f, bq = 0.0f;
  for (int v = v0 + g; v < v1; v += groups) {
    float t = x[(long long)v * C + c];
    a += t; bq += t * t;
  }
  s1[tid] = a; s2[tid] = bq;
  __syncthreads();
  if (g == 0) {
    for (int gg = 1; gg < groups; ++gg) { a += s1[gg * C + c]; bq += s2[gg * C + c]; }
    atomicAdd(&sum[c], a);
    atomicAdd(&sumsq[c], bq);
  }
}

// ---------------------------------------------------------------------------
__global__ void bn_relu_k(const float* __restrict__ x, const float* __restrict__ sum,
                          const float* __restrict__ sumsq, const float* __restrict__ gamma,
                          const float* __restrict__ beta, float invN,
                          _Float16* __restrict__ out, long long n, int C) {
  long long i = (long long)blockIdx.x * 256 + threadIdx.x;
  if (i >= n) return;
  int c = (int)(i % C);
  float m   = sum[c] * invN;
  float var = sumsq[c] * invN - m * m;
  float sc  = __frsqrt_rn(var + 1e-5f) * gamma[c];
  float y   = (x[i] - m) * sc + beta[c];
  out[i] = (_Float16)fmaxf(y, 0.0f);
}

// BN + ReLU + 2x2x2 maxpool fused (relu commutes with max).
__global__ void bn_relu_pool_k(const float* __restrict__ x, const float* __restrict__ sum,
                               const float* __restrict__ sumsq, const float* __restrict__ gamma,
                               const float* __restrict__ beta, float invN,
                               _Float16* __restrict__ out, int D, int C) {
  const int Do = D >> 1;
  long long n = 2LL * Do * Do * Do * C;
  long long i = (long long)blockIdx.x * 256 + threadIdx.x;
  if (i >= n) return;
  int c = (int)(i % C);
  long long v = i / C;
  int zo = (int)(v % Do); v /= Do;
  int yo = (int)(v % Do); v /= Do;
  int xo = (int)(v % Do); int b = (int)(v / Do);
  float m   = sum[c] * invN;
  float var = sumsq[c] * invN - m * m;
  float sc  = __frsqrt_rn(var + 1e-5f) * gamma[c];
  float bb  = beta[c] - m * sc;
  float best = -3.0e38f;
#pragma unroll
  for (int dx = 0; dx < 2; ++dx)
#pragma unroll
    for (int dy = 0; dy < 2; ++dy)
#pragma unroll
      for (int dz = 0; dz < 2; ++dz) {
        long long idx = ((((long long)b * D + (2 * xo + dx)) * D + (2 * yo + dy)) * D
                         + (2 * zo + dz)) * C + c;
        best = fmaxf(best, x[idx] * sc + bb);
      }
  out[i] = (_Float16)fmaxf(best, 0.0f);
}

// BN (no relu) + global average over 8^3 voxels (affine commutes with mean).
__global__ void bn_avg_k(const float* __restrict__ x, const float* __restrict__ sum,
                         const float* __restrict__ sumsq, const float* __restrict__ gamma,
                         const float* __restrict__ beta, float invN,
                         float* __restrict__ out, int C) {
  int i = blockIdx.x * 256 + threadIdx.x;
  if (i >= 2 * C) return;
  int c = i % C, b = i / C;
  float m   = sum[c] * invN;
  float var = sumsq[c] * invN - m * m;
  float sc  = __frsqrt_rn(var + 1e-5f) * gamma[c];
  float bb  = beta[c] - m * sc;
  float a = 0.0f;
  const float* p = x + b * 512 * C;
  for (int v = 0; v < 512; ++v) a += p[v * C + c];
  out[i] = (a * (1.0f / 512.0f)) * sc + bb;
}

// out[b,j] = sum_c h[b,c] * wl[c,j] + bl[j]
__global__ void linear_k(const float* __restrict__ h, const float* __restrict__ wl,
                         const float* __restrict__ bl, float* __restrict__ out) {
  int i = blockIdx.x * 256 + threadIdx.x;
  if (i >= 512) return;
  int j = i % 256, b = i / 256;
  float a = bl[j];
  for (int c = 0; c < 256; ++c) a += h[b * 256 + c] * wl[c * 256 + j];
  out[i] = a;
}

// ---------------------------------------------------------------------------
extern "C" void kernel_launch(void* const* d_in, const int* in_sizes, int n_in,
                              void* d_out, int out_size, void* d_ws, size_t ws_size,
                              hipStream_t stream) {
  (void)in_sizes; (void)n_in; (void)out_size; (void)ws_size;

  static const int CI[8] = {3, 64, 64, 128, 128, 256, 256, 256};
  static const int CO[8] = {64, 64, 128, 128, 256, 256, 256, 256};
  static const int DD[8] = {64, 64, 32, 32, 16, 16, 16, 8};

  const float* xin = (const float*)d_in[0];
  const float* w[8]; const float* g[8]; const float* be[8];
  for (int i = 0; i < 8; ++i) {
    w[i]  = (const float*)d_in[1 + 3 * i];
    g[i]  = (const float*)d_in[2 + 3 * i];
    be[i] = (const float*)d_in[3 + 3 * i];
  }
  const float* wl = (const float*)d_in[25];
  const float* bl = (const float*)d_in[26];
  float* out = (float*)d_out;

  // ---- workspace carve-up (256B aligned) ----------------------------------
  char* ws = (char*)d_ws;
  size_t off = 0;
  auto carve = [&](size_t bytes) -> char* {
    char* p = ws + off;
    off = (off + bytes + 255) & ~(size_t)255;
    return p;
  };
  float*    convb = (float*)   carve(134217728);  // f32 conv out, max 2*64^3*64
  _Float16* actA  = (_Float16*)carve(67108864);   // f16 act ping
  _Float16* actB  = (_Float16*)carve(67108864);   // f16 act pong
  _Float16* wf16  = (_Float16*)carve(13944960);   // all conv weights in f16
  _Float16* x16   = (_Float16*)carve(3145728);    // f16 input (Cin=3)
  float*    stats = (float*)   carve(2048);       // sum[256] ++ sumsq[256]
  float*    avg   = (float*)   carve(2048);       // [2,256] pooled features

  _Float16* w16[8];
  {
    size_t o = 0;
    for (int i = 0; i < 8; ++i) {
      w16[i] = wf16 + o;
      o += (size_t)27 * CI[i] * CO[i];
    }
  }

  // ---- convert input + weights to f16 -------------------------------------
  {
    int n = 2 * 64 * 64 * 64 * 3;
    f32_to_f16_k<<<dim3((n + 255) / 256), 256, 0, stream>>>(xin, x16, n);
    for (int i = 0; i < 8; ++i) {
      int wn = 27 * CI[i] * CO[i];
      f32_to_f16_k<<<dim3((wn + 255) / 256), 256, 0, stream>>>(w[i], w16[i], wn);
    }
  }

  // ---- layer drivers ------------------------------------------------------
  auto launch_conv = [&](const _Float16* in, int li) {
    const int D = DD[li];
    dim3 grid((D + 15) / 16, D * D, 2);
    switch (li) {
      case 0: conv3d_wmma_k<3, 64>   <<<grid, 256, 0, stream>>>(in, w16[0], convb, D); break;
      case 1: conv3d_wmma_k<64, 64>  <<<grid, 256, 0, stream>>>(in, w16[1], convb, D); break;
      case 2: conv3d_wmma_k<64, 128> <<<grid, 256, 0, stream>>>(in, w16[2], convb, D); break;
      case 3: conv3d_wmma_k<128, 128><<<grid, 256, 0, stream>>>(in, w16[3], convb, D); break;
      case 4: conv3d_wmma_k<128, 256><<<grid, 256, 0, stream>>>(in, w16[4], convb, D); break;
      case 5: conv3d_wmma_k<256, 256><<<grid, 256, 0, stream>>>(in, w16[5], convb, D); break;
      case 6: conv3d_wmma_k<256, 256><<<grid, 256, 0, stream>>>(in, w16[6], convb, D); break;
      case 7: conv3d_wmma_k<256, 256><<<grid, 256, 0, stream>>>(in, w16[7], convb, D); break;
    }
  };
  auto conv_stats = [&](const _Float16* in, int li) {
    launch_conv(in, li);
    zero_f32_k<<<2, 256, 0, stream>>>(stats, 512);
    int nvox = 2 * DD[li] * DD[li] * DD[li];
    channel_stats_k<<<256, 256, 0, stream>>>(convb, stats, stats + 256, nvox, CO[li]);
  };
  auto bn_relu = [&](int li, _Float16* dst) {
    int D = DD[li], Cout = CO[li];
    long long n = 2LL * D * D * D * Cout;
    float invN = 1.0f / (float)(2LL * D * D * D);
    bn_relu_k<<<dim3((unsigned)((n + 255) / 256)), 256, 0, stream>>>(
        convb, stats, stats + 256, g[li], be[li], invN, dst, n, Cout);
  };
  auto bn_relu_pool = [&](int li, _Float16* dst) {
    int D = DD[li], Cout = CO[li], Do = D / 2;
    long long n = 2LL * Do * Do * Do * Cout;
    float invN = 1.0f / (float)(2LL * D * D * D);
    bn_relu_pool_k<<<dim3((unsigned)((n + 255) / 256)), 256, 0, stream>>>(
        convb, stats, stats + 256, g[li], be[li], invN, dst, D, Cout);
  };

  conv_stats(x16,  0); bn_relu(0, actA);       // 64^3 x 64
  conv_stats(actA, 1); bn_relu_pool(1, actB);  // -> 32^3 x 64
  conv_stats(actB, 2); bn_relu(2, actA);       // 32^3 x 128
  conv_stats(actA, 3); bn_relu_pool(3, actB);  // -> 16^3 x 128
  conv_stats(actB, 4); bn_relu(4, actA);       // 16^3 x 256
  conv_stats(actA, 5); bn_relu(5, actB);       // 16^3 x 256
  conv_stats(actB, 6); bn_relu_pool(6, actA);  // -> 8^3 x 256
  conv_stats(actA, 7);                         // 8^3 x 256, BN no relu

  {
    float invN = 1.0f / (float)(2LL * 8 * 8 * 8);
    bn_avg_k<<<2, 256, 0, stream>>>(convb, stats, stats + 256, g[7], be[7], invN, avg, 256);
  }
  linear_k<<<2, 256, 0, stream>>>(avg, wl, bl, out);
}
